// AutoEncoderTopK_4664334484114
// MI455X (gfx1250) — matrix operand order repaired and checked by reading, not compiled, not run on hardware
//
#include <hip/hip_runtime.h>

typedef __bf16 bf16;
typedef __attribute__((ext_vector_type(16))) __bf16 v16bf;
typedef __attribute__((ext_vector_type(8)))  __bf16 v8bf;
typedef __attribute__((ext_vector_type(8)))  float  v8f;
typedef unsigned int __attribute__((ext_vector_type(4))) v4u;
typedef int __attribute__((ext_vector_type(4))) v4i;
typedef int __attribute__((ext_vector_type(8))) v8i;

#define B_      8192
#define D_      768
#define F_      24576
#define K_      64
#define NCHUNK  1024
#define NCHUNKS (F_ / NCHUNK)   // 24
#define CAND    (NCHUNKS * K_)  // 1536

// -------------------- helpers --------------------

__device__ __forceinline__ unsigned long long shflxor_u64(unsigned long long v, int m) {
  int lo = (int)(unsigned)(v & 0xffffffffull);
  int hi = (int)(unsigned)(v >> 32);
  lo = __shfl_xor(lo, m, 32);
  hi = __shfl_xor(hi, m, 32);
  return ((unsigned long long)(unsigned)hi << 32) | (unsigned)lo;
}

// -------------------- prep kernels --------------------

// xc[b,d] = bf16(x[b,d] - b_dec[d])
__global__ __launch_bounds__(256) void k_prep_x(const float* __restrict__ x,
                                                const float* __restrict__ b_dec,
                                                bf16* __restrict__ xc) {
  int i = blockIdx.x * 256 + threadIdx.x;      // < B_*D_ = 6.29M
  int d = i - (i / D_) * D_;
  xc[i] = (bf16)(x[i] - b_dec[d]);
}

// wb = bf16(W_enc), flat
__global__ __launch_bounds__(256) void k_prep_wenc(const float* __restrict__ w,
                                                   bf16* __restrict__ wb) {
  size_t i = (size_t)blockIdx.x * 256 + threadIdx.x;
  wb[i] = (bf16)w[i];
}

// W_decT[f,d] = bf16(W_dec[d,f]) — tiled 32x32 transpose via LDS
__global__ __launch_bounds__(256) void k_transpose_wdec(const float* __restrict__ wdec,
                                                        bf16* __restrict__ wdecT) {
  __shared__ float tile[32][33];
  const int tx = threadIdx.x, ty = threadIdx.y;     // block (32, 8)
  const int f0 = blockIdx.x * 32;                   // grid.x = F_/32
  const int d0 = blockIdx.y * 32;                   // grid.y = D_/32
  #pragma unroll
  for (int i = 0; i < 4; ++i)
    tile[ty + 8 * i][tx] = wdec[(size_t)(d0 + ty + 8 * i) * F_ + f0 + tx];
  __syncthreads();
  #pragma unroll
  for (int i = 0; i < 4; ++i)
    wdecT[(size_t)(f0 + ty + 8 * i) * D_ + d0 + tx] = (bf16)tile[tx][ty + 8 * i];
}

// -------------------- encoder GEMM + fused chunk top-k --------------------
// grid: (B_/16, NCHUNKS), block: 256 (8 waves).
// A tile (16 x 768 bf16) staged into LDS by the Tensor Data Mover (one
// tensor_load_to_lds issued by wave 0, s_wait_tensorcnt, then barrier).
// Workgroup computes preact tile [16 rows x 1024 cols] with WMMA bf16,
// applies +b_enc and ReLU into LDS, then each wave extracts top-64 of its
// 2 rows for this chunk and emits candidates.
__global__ __launch_bounds__(256) void k_encode(const bf16* __restrict__ xc,
                                                const bf16* __restrict__ wenc,
                                                const float* __restrict__ b_enc,
                                                float* __restrict__ cand_val,
                                                int* __restrict__ cand_idx) {
  __shared__ bf16  lds_a[16 * D_];        // 24 KB: staged x rows
  __shared__ float lds_p[16 * NCHUNK];    // 64 KB: post-ReLU preact tile

  const int tid  = threadIdx.x;
  const int lane = tid & 31;
  const int wv   = tid >> 5;
  const int m0   = blockIdx.x * 16;
  const int chunk = blockIdx.y;
  const int f0   = chunk * NCHUNK;

#if __has_builtin(__builtin_amdgcn_tensor_load_to_lds) && \
    __has_builtin(__builtin_amdgcn_s_wait_tensorcnt)
  // --- TDM path: one 2D tensor descriptor moves the whole 16x768 bf16 tile.
  if (wv == 0) {
    const unsigned long long ga =
        (unsigned long long)(uintptr_t)(xc + (size_t)m0 * D_);
    // Flat LDS aperture: low 32 bits of the generic address == LDS byte addr.
    const unsigned lds0 = (unsigned)(uintptr_t)(&lds_a[0]);

    // D# group 0: count=1 | lds_addr | global_addr[56:0] | type=2
    v4u g0;
    g0.x = 1u;
    g0.y = lds0;
    g0.z = (unsigned)(ga & 0xffffffffull);
    g0.w = (unsigned)((ga >> 32) & 0x1ffffffull) | (2u << 30);

    // D# group 1: data_size=1 (2B elems); tensor 768x16; tile 768x16;
    //             tensor_dim0_stride = 768 elements.
    const unsigned dim0 = (unsigned)D_;   // elements along d
    const unsigned dim1 = 16u;            // rows
    v8i g1 = {
        (int)(1u << 16),                                    // data_size=2B
        (int)((dim0 & 0xffffu) << 16),                      // tensor_dim0 lo16
        (int)(((dim0 >> 16) & 0xffffu) | ((dim1 & 0xffffu) << 16)), // dim0 hi | dim1 lo
        (int)(((dim1 >> 16) & 0xffffu) | (dim0 << 16)),     // dim1 hi | tile_dim0=768
        (int)dim1,                                          // tile_dim1=16, tile_dim2=0
        (int)dim0,                                          // dim0_stride lo32 = 768
        0, 0 };
    v4i gz4 = {0, 0, 0, 0};
    v8i gz8 = {0, 0, 0, 0, 0, 0, 0, 0};
    __builtin_amdgcn_tensor_load_to_lds(g0, g1, gz4, gz4, gz8, 0);
    __builtin_amdgcn_s_wait_tensorcnt((short)0);
  }
#else
  // --- Fallback: coalesced dword copy.
  {
    const unsigned* src = (const unsigned*)(xc + (size_t)m0 * D_);
    unsigned* dst = (unsigned*)lds_a;
    for (int i = tid; i < (16 * D_) / 2; i += 256) dst[i] = src[i];
  }
#endif
  __syncthreads();

  const int half = lane >> 4;   // which K-half this lane holds
  const int l15  = lane & 15;
  const int nwave = f0 + wv * 128;

  v8f acc[8];
  v8f zero = {0.f, 0.f, 0.f, 0.f, 0.f, 0.f, 0.f, 0.f};
  #pragma unroll
  for (int t = 0; t < 8; ++t) acc[t] = zero;

  // K loop over D_ in steps of 32 (v_wmma_f32_16x16x32_bf16)
  for (int ks = 0; ks < D_ / 32; ++ks) {
    const int kk = ks * 32;
    // A fragment (16x32 bf16): lane holds row M=l15,
    // 16 contiguous K values starting at kk + 16*half
    const bf16* pa = &lds_a[l15 * D_ + kk + half * 16];
    v8bf alo = *(const v8bf*)(pa);
    v8bf ahi = *(const v8bf*)(pa + 8);
    v16bf afrag = __builtin_shufflevector(alo, ahi,
        0, 1, 2, 3, 4, 5, 6, 7, 8, 9, 10, 11, 12, 13, 14, 15);

    #pragma unroll
    for (int t = 0; t < 8; ++t) {
      // B fragment (32x16 bf16): lane holds col N=l15 (= W_enc row f),
      // 16 contiguous K values starting at kk + 16*half.
      const size_t f = (size_t)(nwave + t * 16 + l15);
      const bf16* pb = wenc + f * D_ + kk + half * 16;
      __builtin_prefetch(pb + 32, 0, 0);   // next k-step -> global_prefetch_b8
      const v16bf bfrag = *(const v16bf*)pb;
      acc[t] = __builtin_amdgcn_wmma_f32_16x16x32_bf16(
          false, afrag, false, bfrag, (short)0, acc[t], false, false);
    }
  }

  // Epilogue: +b_enc, ReLU, write tile to LDS.
  // C layout: VGPR i, lane L -> M = i + 8*(L>=16), N = L&15.
  #pragma unroll
  for (int t = 0; t < 8; ++t) {
    const int col = wv * 128 + t * 16 + l15;
    const float be = b_enc[f0 + col];
    #pragma unroll
    for (int i = 0; i < 8; ++i) {
      const int row = i + half * 8;
      float v = acc[t][i] + be;
      lds_p[row * NCHUNK + col] = v > 0.f ? v : 0.f;
    }
  }
  __syncthreads();

  // Chunk-local top-64 per row; wave wv owns rows 2wv, 2wv+1.
  for (int rr = 0; rr < 2; ++rr) {
    const int row = wv * 2 + rr;
    unsigned long long pk[32];
    #pragma unroll
    for (int j = 0; j < 32; ++j) {
      const int col = lane + 32 * j;
      const float v = lds_p[row * NCHUNK + col];   // v >= 0 -> bits monotonic
      pk[j] = ((unsigned long long)__float_as_uint(v) << 32) | (unsigned)col;
    }
    const size_t base = (size_t)(m0 + row) * CAND + (size_t)chunk * K_;
    for (int k = 0; k < K_; ++k) {
      unsigned long long best = pk[0];
      #pragma unroll
      for (int j = 1; j < 32; ++j) best = pk[j] > best ? pk[j] : best;
      #pragma unroll
      for (int s = 16; s > 0; s >>= 1) {
        unsigned long long o = shflxor_u64(best, s);
        best = o > best ? o : best;
      }
      const unsigned col = (unsigned)best & 1023u;
      if (lane == 0) {
        cand_val[base + k] = __uint_as_float((unsigned)(best >> 32));
        cand_idx[base + k] = f0 + (int)col;
      }
      // Owner lane clears the winner (to a value that can't beat real entries).
      if ((col & 31u) == (unsigned)lane) {
        const int oj = (int)(col >> 5);
        #pragma unroll
        for (int j = 0; j < 32; ++j)
          if (j == oj) pk[j] = (unsigned long long)col;
      }
    }
  }
}

// -------------------- merge top-k + sparse decode --------------------
// grid: B_ blocks, block: 256. Merge 1536 candidates -> top 64, then
// x_hat[b,:] = sum_k v_k * W_decT[idx_k,:] + b_dec.
__global__ __launch_bounds__(256) void k_decode(const float* __restrict__ cand_val,
                                                const int* __restrict__ cand_idx,
                                                const bf16* __restrict__ wdecT,
                                                const float* __restrict__ b_dec,
                                                float* __restrict__ out) {
  __shared__ unsigned cv[CAND];
  __shared__ int      ci[CAND];
  __shared__ unsigned long long red[8];
  __shared__ float topv[K_];
  __shared__ int   topi[K_];

  const int tid = threadIdx.x, lane = tid & 31, wv = tid >> 5;
  const int b = blockIdx.x;
  const size_t base = (size_t)b * CAND;

  for (int i = tid; i < CAND; i += 256) {
    cv[i] = __float_as_uint(cand_val[base + i]);  // all >= 0
    ci[i] = cand_idx[base + i];
  }
  __syncthreads();

  for (int k = 0; k < K_; ++k) {
    unsigned long long best = 0ull;
    #pragma unroll
    for (int j = 0; j < 6; ++j) {
      const int p = tid * 6 + j;
      unsigned long long e = ((unsigned long long)cv[p] << 32) | (unsigned)p;
      best = e > best ? e : best;
    }
    #pragma unroll
    for (int s = 16; s > 0; s >>= 1) {
      unsigned long long o = shflxor_u64(best, s);
      best = o > best ? o : best;
    }
    if (lane == 0) red[wv] = best;
    __syncthreads();
    if (tid == 0) {
      unsigned long long bb = red[0];
      #pragma unroll
      for (int w = 1; w < 8; ++w) bb = red[w] > bb ? red[w] : bb;
      const unsigned p = (unsigned)bb & 0xffffu;
      topv[k] = __uint_as_float((unsigned)(bb >> 32));
      topi[k] = ci[p];
      cv[p] = 0u;   // re-picks only ever tie at value 0 -> contributes nothing
    }
    __syncthreads();
  }

  // Sparse decode: thread tid accumulates d = tid, tid+256, tid+512.
  float a0 = 0.f, a1 = 0.f, a2 = 0.f;
  const int d0 = tid, d1 = tid + 256, d2 = tid + 512;
  for (int k = 0; k < K_; ++k) {
    const float v = topv[k];
    const bf16* wrow = wdecT + (size_t)topi[k] * D_;
    a0 += v * (float)wrow[d0];
    a1 += v * (float)wrow[d1];
    a2 += v * (float)wrow[d2];
  }
  const size_t ob = (size_t)b * D_;
  out[ob + d0] = a0 + b_dec[d0];
  out[ob + d1] = a1 + b_dec[d1];
  out[ob + d2] = a2 + b_dec[d2];
}

// -------------------- launcher --------------------

extern "C" void kernel_launch(void* const* d_in, const int* in_sizes, int n_in,
                              void* d_out, int out_size, void* d_ws, size_t ws_size,
                              hipStream_t stream) {
  const float* x     = (const float*)d_in[0];   // [B, D]
  const float* W_enc = (const float*)d_in[1];   // [F, D]
  const float* b_enc = (const float*)d_in[2];   // [F]
  const float* W_dec = (const float*)d_in[3];   // [D, F]
  const float* b_dec = (const float*)d_in[4];   // [D]
  float* out = (float*)d_out;                   // [B, D]

  char* ws = (char*)d_ws;
  size_t off = 0;
  bf16* xc     = (bf16*)(ws + off); off += (size_t)B_ * D_ * sizeof(bf16);   // 12.6 MB
  bf16* wencb  = (bf16*)(ws + off); off += (size_t)F_ * D_ * sizeof(bf16);   // 37.7 MB
  bf16* wdecT  = (bf16*)(ws + off); off += (size_t)F_ * D_ * sizeof(bf16);   // 37.7 MB
  float* cval  = (float*)(ws + off); off += (size_t)B_ * CAND * sizeof(float); // 50.3 MB
  int*   cidx  = (int*)(ws + off);   off += (size_t)B_ * CAND * sizeof(int);   // 50.3 MB
  (void)ws_size; (void)in_sizes; (void)n_in; (void)out_size;

  k_prep_x<<<(B_ * D_) / 256, 256, 0, stream>>>(x, b_dec, xc);
  k_prep_wenc<<<(F_ * D_) / 256, 256, 0, stream>>>(W_enc, wencb);
  k_transpose_wdec<<<dim3(F_ / 32, D_ / 32), dim3(32, 8), 0, stream>>>(W_dec, wdecT);

  k_encode<<<dim3(B_ / 16, NCHUNKS), 256, 0, stream>>>(xc, wencb, b_enc, cval, cidx);

  k_decode<<<B_, 256, 0, stream>>>(cval, cidx, wdecT, b_dec, out);
}
